// Actor_69071664054391
// MI455X (gfx1250) — matrix operand order repaired and checked by reading, not compiled, not run on hardware
//
#include <hip/hip_runtime.h>
#include <hip/hip_bf16.h>

typedef _Float16 half_t;
typedef __attribute__((ext_vector_type(16))) _Float16 v16h;
typedef __attribute__((ext_vector_type(8)))  _Float16 v8h;
typedef __attribute__((ext_vector_type(8)))  float    v8f;

#define HID 64

__device__ __forceinline__ int imin(int a, int b) { return a < b ? a : b; }

// ---------------------------------------------------------------------------
// Zero a region (works for f32 0.0 and u32 key 0)
// ---------------------------------------------------------------------------
__global__ void k_zero(unsigned* p, int n) {
    int i = blockIdx.x * blockDim.x + threadIdx.x;
    if (i < n) p[i] = 0u;
}

// ---------------------------------------------------------------------------
// Swizzle a f32 [kblocks*32, 64] weight matrix into the WMMA B-operand lane
// layout (f16), so each lane's v16h is one contiguous 32-byte load.
// dst index d = ((ct*kblocks + kb)*32 + lane)*16 + j
//   k = kb*32 + (lane>>4)*8 + (j&7) + (j>>3)*16 ;  c = ct*16 + (lane&15)
// ---------------------------------------------------------------------------
__global__ void k_swizzleB(const float* __restrict__ W, half_t* __restrict__ dst,
                           int kblocks) {
    int d = blockIdx.x * blockDim.x + threadIdx.x;
    int total = kblocks * 2048;
    if (d >= total) return;
    int j    = d & 15;
    int lane = (d >> 4) & 31;
    int rem  = d >> 9;
    int kb   = rem % kblocks;
    int ct   = rem / kblocks;
    int k = kb * 32 + (lane >> 4) * 8 + (j & 7) + (j >> 3) * 16;
    int c = ct * 16 + (lane & 15);
    dst[d] = (half_t)W[k * HID + c];
}

// ---------------------------------------------------------------------------
// Node encoder: n = relu(x @ Wn + bn), stored directly as f16 (WMMA A input)
// ---------------------------------------------------------------------------
__global__ void k_enc_nodes(const float* __restrict__ x, const float* __restrict__ Wn,
                            const float* __restrict__ bn, half_t* __restrict__ nh, int N) {
    int i = blockIdx.x * blockDim.x + threadIdx.x;
    if (i >= N * HID) return;
    int node = i >> 6, f = i & 63;
    float v = x[node * 3 + 0] * Wn[0 * HID + f]
            + x[node * 3 + 1] * Wn[1 * HID + f]
            + x[node * 3 + 2] * Wn[2 * HID + f] + bn[f];
    nh[i] = (half_t)fmaxf(v, 0.0f);
}

// ---------------------------------------------------------------------------
// Node-level triple GEMM: O{0,1,2}[N,64] = Ah[N,64] @ B{0,1,2}[64,64]
// One wave per 16x16 output tile; K=64 via two v_wmma_f32_16x16x32_f16.
// blockDim = 128 (4 waves); grid.y = 12 = (which weight)*4 + col tile.
// ---------------------------------------------------------------------------
__global__ void k_node_gemm3(const half_t* __restrict__ Ah,
                             const half_t* __restrict__ B0,
                             const half_t* __restrict__ B1,
                             const half_t* __restrict__ B2,
                             float* __restrict__ O0, float* __restrict__ O1,
                             float* __restrict__ O2, int N) {
    int waveid = threadIdx.x >> 5;
    int lane   = threadIdx.x & 31;
    int tile   = blockIdx.x * 4 + waveid;
    int row0   = tile * 16;
    if (row0 >= N) return;                        // wave-uniform

    int w  = blockIdx.y >> 2;
    int ct = blockIdx.y & 3;
    const half_t* B = (w == 0) ? B0 : (w == 1 ? B1 : B2);
    float*        O = (w == 0) ? O0 : (w == 1 ? O1 : O2);

    int r  = lane & 15;
    int hi = lane >> 4;
    int row = imin(row0 + r, N - 1);
    const half_t* rp = Ah + row * HID;
    int kb = hi * 8;

    // A operand, ISA 16-bit 16x32 layout: lane holds K = kb..kb+7, kb+16..kb+23
    v8h lo0 = *(const v8h*)(rp + kb);
    v8h hi0 = *(const v8h*)(rp + kb + 16);
    v8h lo1 = *(const v8h*)(rp + 32 + kb);
    v8h hi1 = *(const v8h*)(rp + 32 + kb + 16);
    v16h A0, A1;
#pragma unroll
    for (int j = 0; j < 8; ++j) {
        A0[j] = lo0[j]; A0[8 + j] = hi0[j];
        A1[j] = lo1[j]; A1[8 + j] = hi1[j];
    }

    // Pre-swizzled B: one contiguous 32B load per lane per K-block
    v16h Bv0 = *(const v16h*)(B + ((ct * 2 + 0) * 32 + lane) * 16);
    v16h Bv1 = *(const v16h*)(B + ((ct * 2 + 1) * 32 + lane) * 16);

    v8f acc = {};
    acc = __builtin_amdgcn_wmma_f32_16x16x32_f16(false, A0, false, Bv0,
                                                 (short)0, acc, false, false);
    acc = __builtin_amdgcn_wmma_f32_16x16x32_f16(false, A1, false, Bv1,
                                                 (short)0, acc, false, false);

    int col = ct * 16 + (lane & 15);
    if (row0 + 16 <= N) {                         // wave-uniform fast path
#pragma unroll
        for (int v = 0; v < 8; ++v)
            O[(row0 + v + hi * 8) * HID + col] = acc[v];
    } else {
#pragma unroll
        for (int v = 0; v < 8; ++v) {
            int rr = row0 + v + hi * 8;           // C/D layout: M = v + 8*(lane>>4)
            if (rr < N) O[rr * HID + col] = acc[v];
        }
    }
}

// ---------------------------------------------------------------------------
// conv1 edge pipeline: per wave of 16 edges
//   e = relu(edge_attr @ We + be)        (each lane computes its A-slot feats)
//   msg = e @ W1_edge (4x WMMA)  + t_{etype}[src]   -> atomicAdd into agg[dst]
// W1_edge B operand staged block-wide into LDS via async load (ASYNCcnt path).
// ---------------------------------------------------------------------------
__global__ void k_conv1_edge(const float* __restrict__ ea,
                             const int* __restrict__ eidx,
                             const int* __restrict__ etype,
                             const float* __restrict__ We,
                             const float* __restrict__ be,
                             const half_t* __restrict__ W1eB,
                             const float* __restrict__ t0,
                             const float* __restrict__ t1,
                             float* __restrict__ agg, int E) {
    __shared__ half_t sB[2048];                   // 4 KB: swizzled W1_edge [32,64]

    // Block-wide async stage: 256 threads x 16 B = 4096 B
    {
        unsigned lds_off = (unsigned)(uintptr_t)(&sB[threadIdx.x * 8]);
        const half_t* g  = W1eB + threadIdx.x * 8;
        asm volatile("global_load_async_to_lds_b128 %0, %1, off"
                     :: "v"(lds_off), "v"(g) : "memory");
        asm volatile("s_wait_asynccnt 0x0" ::: "memory");
    }
    __syncthreads();

    int wave = (blockIdx.x * blockDim.x + threadIdx.x) >> 5;
    int lane = threadIdx.x & 31;
    int eb   = wave * 16;
    if (eb >= E) return;                          // wave-uniform

    int r  = lane & 15;
    int hi = lane >> 4;
    int eA = imin(eb + r, E - 1);
    float a0f = ea[eA * 2 + 0], a1f = ea[eA * 2 + 1];
    int kb = hi * 8;

    v16h A;
#pragma unroll
    for (int j = 0; j < 8; ++j) {
        int f0 = kb + j, f1 = kb + 16 + j;
        float v0 = fmaxf(a0f * We[f0] + a1f * We[32 + f0] + be[f0], 0.0f);
        float v1 = fmaxf(a0f * We[f1] + a1f * We[32 + f1] + be[f1], 0.0f);
        A[j] = (half_t)v0; A[8 + j] = (half_t)v1;
    }

    bool full = (eb + 16 <= E);                   // wave-uniform
    int ms[8], md[8], mt[8];                      // per-lane edge metadata (C rows)
#pragma unroll
    for (int v = 0; v < 8; ++v) {
        int ec = imin(eb + v + hi * 8, E - 1);
        ms[v] = eidx[ec];
        md[v] = eidx[E + ec];
        mt[v] = etype[ec];
    }

#pragma unroll
    for (int ct = 0; ct < 4; ++ct) {
        v16h B = *(const v16h*)(sB + (ct * 32 + lane) * 16);   // ds_load path
        v8f acc = {};
        acc = __builtin_amdgcn_wmma_f32_16x16x32_f16(false, A, false, B,
                                                     (short)0, acc, false, false);
        int col = ct * 16 + (lane & 15);
        if (full) {
#pragma unroll
            for (int v = 0; v < 8; ++v) {
                const float* tsel = mt[v] ? t1 : t0;
                float val = acc[v] + tsel[ms[v] * HID + col];
                atomicAdd(&agg[md[v] * HID + col], val);
            }
        } else {
#pragma unroll
            for (int v = 0; v < 8; ++v) {
                int e = eb + v + hi * 8;
                if (e < E) {
                    const float* tsel = mt[v] ? t1 : t0;
                    float val = acc[v] + tsel[ms[v] * HID + col];
                    atomicAdd(&agg[md[v] * HID + col], val);
                }
            }
        }
    }
}

// ---------------------------------------------------------------------------
// h = relu(agg1 + nroot + b1), stored as f16 for conv2 GEMMs
// ---------------------------------------------------------------------------
__global__ void k_make_h(const float* __restrict__ agg1, const float* __restrict__ nroot,
                         const float* __restrict__ b1, half_t* __restrict__ hh, int N) {
    int i = blockIdx.x * blockDim.x + threadIdx.x;
    if (i >= N * HID) return;
    int f = i & 63;
    hh[i] = (half_t)fmaxf(agg1[i] + nroot[i] + b1[f], 0.0f);
}

// ---------------------------------------------------------------------------
// conv2 edge: scatter-max of u_{etype}[src] into dst via monotone u32 keys
// ---------------------------------------------------------------------------
__device__ __forceinline__ unsigned f2key(float v) {
    unsigned b = __float_as_uint(v);
    return (b & 0x80000000u) ? ~b : (b | 0x80000000u);
}

__global__ void k_conv2_edge(const int* __restrict__ eidx, const int* __restrict__ etype,
                             const float* __restrict__ u0, const float* __restrict__ u1,
                             unsigned* __restrict__ aggk, int E) {
    int i = blockIdx.x * blockDim.x + threadIdx.x;
    if (i >= E * 16) return;
    int e = i >> 4, q = i & 15;
    int s = eidx[e], d = eidx[E + e];
    const float* u = etype[e] ? u1 : u0;
    int base = q * 4;
    float4 va = *(const float4*)(u + s * HID + base);          // global_load_b128
    atomicMax(&aggk[d * HID + base + 0], f2key(va.x));
    atomicMax(&aggk[d * HID + base + 1], f2key(va.y));
    atomicMax(&aggk[d * HID + base + 2], f2key(va.z));
    atomicMax(&aggk[d * HID + base + 3], f2key(va.w));
}

// ---------------------------------------------------------------------------
// Final: h2 = relu(decode(agg2) + hroot + b2); out = tanh(h2 . Wc + bc) * 5
// ---------------------------------------------------------------------------
__global__ void k_final(const unsigned* __restrict__ aggk, const float* __restrict__ hroot,
                        const float* __restrict__ b2, const float* __restrict__ Wc,
                        const float* __restrict__ bc, float* __restrict__ out, int N) {
    __shared__ float sdata[HID];
    int node = blockIdx.x;
    int f    = threadIdx.x;
    unsigned k = aggk[node * HID + f];
    unsigned b = (k & 0x80000000u) ? (k & 0x7fffffffu) : ~k;
    float a = __uint_as_float(b);
    if (!isfinite(a)) a = 0.0f;                   // empty segments -> 0
    float h2 = fmaxf(a + hroot[node * HID + f] + b2[f], 0.0f);
    sdata[f] = h2 * Wc[f];
    __syncthreads();
#pragma unroll
    for (int s = 32; s > 0; s >>= 1) {
        if (f < s) sdata[f] += sdata[f + s];
        __syncthreads();
    }
    if (f == 0) out[node] = tanhf(sdata[0] + bc[0]) * 5.0f;
}

// ---------------------------------------------------------------------------
extern "C" void kernel_launch(void* const* d_in, const int* in_sizes, int n_in,
                              void* d_out, int out_size, void* d_ws, size_t ws_size,
                              hipStream_t stream) {
    const float* x   = (const float*)d_in[0];
    const int*   ei  = (const int*)  d_in[1];
    const float* ea  = (const float*)d_in[2];
    const int*   et  = (const int*)  d_in[3];
    const float* Wn  = (const float*)d_in[4];
    const float* bn  = (const float*)d_in[5];
    const float* We  = (const float*)d_in[6];
    const float* be  = (const float*)d_in[7];
    const float* W1r = (const float*)d_in[8];     // [2,64,64]
    const float* W1e = (const float*)d_in[9];     // [32,64]
    const float* W1t = (const float*)d_in[10];    // [64,64]
    const float* b1  = (const float*)d_in[11];
    const float* W2r = (const float*)d_in[12];    // [2,64,64]
    const float* W2t = (const float*)d_in[13];    // [64,64]
    const float* b2  = (const float*)d_in[14];
    const float* Wc  = (const float*)d_in[15];
    const float* bc  = (const float*)d_in[16];
    float* out = (float*)d_out;

    const int N = in_sizes[0] / 3;
    const int E = in_sizes[3];
    const size_t NF = (size_t)N * HID;

    // Workspace layout (float units)
    float*   S0   = (float*)d_ws;                 // t0 / u0
    float*   S1   = S0 + NF;                      // t1 / u1
    float*   SR   = S0 + 2 * NF;                  // nroot / hroot
    float*   AGG  = S0 + 3 * NF;                  // agg1 (f32) then agg2 keys (u32)
    half_t*  NH   = (half_t*)(S0 + 4 * NF);       // n as f16  [N,64]
    half_t*  HH   = (half_t*)(S0 + 4 * NF + NF / 2); // h as f16
    half_t*  WH   = (half_t*)(S0 + 5 * NF);       // swizzled f16 weights
    half_t*  w1eB  = WH;                          // 2048
    half_t*  w1r0B = WH + 2048;                   // 4096 each below
    half_t*  w1r1B = WH + 6144;
    half_t*  w1rtB = WH + 10240;
    half_t*  w2r0B = WH + 14336;
    half_t*  w2r1B = WH + 18432;
    half_t*  w2rtB = WH + 22528;

    const int BT = 256;
    int nfBlocks = (int)((NF + BT - 1) / BT);
    int tiles    = (N + 15) / 16;

    // 0) zero agg1
    k_zero<<<nfBlocks, BT, 0, stream>>>((unsigned*)AGG, (int)NF);

    // 1) swizzle weights -> f16 WMMA-B layout
    k_swizzleB<<<(2048 + BT - 1) / BT, BT, 0, stream>>>(W1e, w1eB, 1);
    k_swizzleB<<<(4096 + BT - 1) / BT, BT, 0, stream>>>(W1r,        w1r0B, 2);
    k_swizzleB<<<(4096 + BT - 1) / BT, BT, 0, stream>>>(W1r + 4096, w1r1B, 2);
    k_swizzleB<<<(4096 + BT - 1) / BT, BT, 0, stream>>>(W1t,        w1rtB, 2);
    k_swizzleB<<<(4096 + BT - 1) / BT, BT, 0, stream>>>(W2r,        w2r0B, 2);
    k_swizzleB<<<(4096 + BT - 1) / BT, BT, 0, stream>>>(W2r + 4096, w2r1B, 2);
    k_swizzleB<<<(4096 + BT - 1) / BT, BT, 0, stream>>>(W2t,        w2rtB, 2);

    // 2) node encoder -> NH (f16)
    k_enc_nodes<<<nfBlocks, BT, 0, stream>>>(x, Wn, bn, NH, N);

    // 3) hoisted relation GEMMs: t0,t1,nroot = n @ {W1_rel0, W1_rel1, W1_root}
    dim3 g3((tiles + 3) / 4, 12);
    k_node_gemm3<<<g3, 128, 0, stream>>>(NH, w1r0B, w1r1B, w1rtB, S0, S1, SR, N);

    // 4) conv1 edge pipeline (edge MLP + WMMA edge-term GEMM + gather + scatter-add)
    int waves1 = (E + 15) / 16;
    k_conv1_edge<<<(waves1 + 7) / 8, 256, 0, stream>>>(ea, ei, et, We, be, w1eB,
                                                       S0, S1, AGG, E);

    // 5) h = relu(agg1 + nroot + b1) -> HH (f16)
    k_make_h<<<nfBlocks, BT, 0, stream>>>(AGG, SR, b1, HH, N);

    // 6) re-zero AGG as u32 max-keys
    k_zero<<<nfBlocks, BT, 0, stream>>>((unsigned*)AGG, (int)NF);

    // 7) conv2 relation GEMMs: u0,u1,hroot = h @ {W2_rel0, W2_rel1, W2_root}
    k_node_gemm3<<<g3, 128, 0, stream>>>(HH, w2r0B, w2r1B, w2rtB, S0, S1, SR, N);

    // 8) conv2 edge scatter-max
    int th2 = E * 16;
    k_conv2_edge<<<(th2 + BT - 1) / BT, BT, 0, stream>>>(ei, et, S0, S1,
                                                         (unsigned*)AGG, E);

    // 9) final readout
    k_final<<<N, HID, 0, stream>>>((unsigned*)AGG, SR, b2, Wc, bc, out, N);
}
